// EGNNScoreNet_31653908971882
// MI455X (gfx1250) — matrix operand order repaired and checked by reading, not compiled, not run on hardware
//
#include <hip/hip_runtime.h>
#include <hip/hip_bf16.h>

// ============================================================================
// EGNN score network for MI455X (gfx1250, wave32, WMMA bf16 16x16x32).
//
//  * ~1.3 TFLOP but ~13 GB activation traffic -> memory bound (~0.6 ms @
//    23.3 TB/s). bf16 WMMA; fused message+coord MLP chain; contiguous
//    99-edge segment reductions (edge list is row-sorted); no atomics.
//  * Weights LDS-resident, stored TRANSPOSED (Wt[n][k], ldk=K+8) so each
//    WMMA B fragment is two ds_load_b128; A fragments likewise two
//    ds_load_b128 from 136-stride staging tiles. K loops fully unrolled.
//  * LN / row dots computed in the WMMA C layout with shfl_xor reductions
//    over the 16-lane halves (wave32-native).
// ============================================================================

typedef __attribute__((ext_vector_type(16))) __bf16 v16bf;
typedef __attribute__((ext_vector_type(8)))  __bf16 v8bf;
typedef __attribute__((ext_vector_type(8)))  float  v8f;

#define B_        64
#define NNODE     100
#define NEDGE     9900
#define NROWS     (B_ * NEDGE)      // 633600 edge rows (divisible by 128)
#define NTILE128  (NROWS / 128)     // 4950
#define NODE_ROWS (B_ * NNODE)      // 6400  (divisible by 128)

__device__ __forceinline__ float silu_f(float v) { return v / (1.f + __expf(-v)); }

__device__ __forceinline__ float rowsum16(float v) {
  v += __shfl_xor(v, 1, 32);
  v += __shfl_xor(v, 2, 32);
  v += __shfl_xor(v, 4, 32);
  v += __shfl_xor(v, 8, 32);
  return v;
}

__device__ __forceinline__ v16bf frag16(v8bf lo, v8bf hi) {
  return __builtin_shufflevector(lo, hi, 0,1,2,3,4,5,6,7,8,9,10,11,12,13,14,15);
}

// pack 4 f32 -> 4 bf16 -> single 8-byte LDS store
union BF4 { __bf16 h[4]; uint2 u; };
__device__ __forceinline__ void st4(__bf16* dst, float a, float b, float c, float d) {
  BF4 t; t.h[0]=(__bf16)a; t.h[1]=(__bf16)b; t.h[2]=(__bf16)c; t.h[3]=(__bf16)d;
  *(uint2*)dst = t.u;
}

// ---- cooperative weight loaders (blockDim.x == 256) ------------------------
// Transposed: global W[k*N+n] (f32) -> LDS Wt[n*ldk+k] (bf16), ldk = K+8.
__device__ __forceinline__ void ldw_bf16_t(const float* __restrict__ s, __bf16* d,
                                           int K, int N, int ldk, int tid) {
  for (int i = tid; i < K * N; i += 256) {
    int k = i / N, n = i - k * N;
    d[n * ldk + k] = (__bf16)s[i];
  }
}
__device__ __forceinline__ void ldw_f32(const float* __restrict__ s, float* d, int n, int tid) {
  for (int i = tid; i < n; i += 256) d[i] = s[i];
}

// ---- WMMA tile GEMM: acc[NT] (16 x NT*16) += A(16xK) * W(KxNT*16) ----------
// A: LDS row-major [16][lda] bf16 (lda = 136 or 200, 16B-aligned rows).
// Wt: LDS transposed [NT*16][ldk] bf16.
template<int NT, int K>
__device__ __forceinline__ void gemm_tile(v8f (&acc)[NT], const __bf16* __restrict__ A, int lda,
                                          const __bf16* __restrict__ Wt, int ldk, int lane) {
  const int m   = lane & 15;
  const int kbA = (lane >> 4) << 3;   // A layout: upper half-wave holds K+8
  const int kbB = (lane >> 4) << 3;   // B layout: upper half-wave holds K+8
#pragma unroll
  for (int kc = 0; kc < K; kc += 32) {
    const __bf16* ap = A + m * lda + kc + kbA;
    v16bf a = frag16(*(const v8bf*)ap, *(const v8bf*)(ap + 16));
#pragma unroll
    for (int t = 0; t < NT; ++t) {
      const __bf16* bp = Wt + (t * 16 + m) * ldk + kc + kbB;
      v16bf bb = frag16(*(const v8bf*)bp, *(const v8bf*)(bp + 16));
      acc[t] = __builtin_amdgcn_wmma_f32_16x16x32_bf16(false, a, false, bb,
                                                       (short)0, acc[t], false, false);
    }
  }
}

template<int NT>
__device__ __forceinline__ void zero_acc(v8f (&acc)[NT]) {
#pragma unroll
  for (int t = 0; t < NT; ++t)
#pragma unroll
    for (int q = 0; q < 8; ++q) acc[t][q] = 0.f;
}

template<int NT>
__device__ __forceinline__ void bias_act(v8f (&acc)[NT], const float* __restrict__ bias,
                                         int lane, bool do_silu) {
#pragma unroll
  for (int t = 0; t < NT; ++t) {
    float bv = bias[(lane & 15) + 16 * t];
#pragma unroll
    for (int j = 0; j < 8; ++j) {
      float v = acc[t][j] + bv;
      acc[t][j] = do_silu ? silu_f(v) : v;
    }
  }
}

// LayerNorm over the NT*16 columns of each row, in C-fragment layout.
template<int NT>
__device__ __forceinline__ void layer_norm(v8f (&acc)[NT], const float* __restrict__ g,
                                           const float* __restrict__ b, int lane) {
  const float invN = 1.f / (float)(NT * 16);
#pragma unroll
  for (int j = 0; j < 8; ++j) {
    float s = 0.f, s2 = 0.f;
#pragma unroll
    for (int t = 0; t < NT; ++t) { float v = acc[t][j]; s += v; s2 += v * v; }
    s = rowsum16(s); s2 = rowsum16(s2);
    float mu = s * invN;
    float rs = rsqrtf(s2 * invN - mu * mu + 1e-5f);
#pragma unroll
    for (int t = 0; t < NT; ++t) {
      int cc = (lane & 15) + 16 * t;
      acc[t][j] = (acc[t][j] - mu) * rs * g[cc] + b[cc];
    }
  }
}

// Store C fragments as bf16 into an LDS tile (row-major, leading dim ld).
template<int NT>
__device__ __forceinline__ void acc_to_lds(const v8f (&acc)[NT], __bf16* T, int ld, int lane) {
  const int rbase = (lane >> 4) << 3;
  const int cbase = lane & 15;
#pragma unroll
  for (int t = 0; t < NT; ++t)
#pragma unroll
    for (int j = 0; j < 8; ++j)
      T[(rbase + j) * ld + cbase + 16 * t] = (__bf16)acc[t][j];
}

// ---- kernel parameter packs ------------------------------------------------
struct MsgP  { const float *Wh,*We,*wd,*b0,*g,*be,*W1,*b1,*W2,*b2,*Wc0,*bc0,*wc1; };
struct NodeP { const float *W0,*b0,*lg,*lb,*W1,*b1,*ng,*nb; };
struct EdgeP { const float *W0a,*W0b,*b0,*lg,*lb,*W1,*b1,*ng,*nb; };
struct OutP  { const float *l1g,*l1b,*W1,*b1,*l2g,*l2b,*W2,*b2,*W3,*b3; };
struct TimeP { const float *W0,*b0,*W1,*b1; const float *tw[8]; const float *tb[8]; };

// ============================================================================
// Fused message MLP + coord MLP over all B*E edge rows.
// msg_in = [h_row(64) | h_col(64) | dist(1) | e_t(128)]; dist handled rank-1.
// ============================================================================
__global__ __launch_bounds__(256) void k_message(
    const float* __restrict__ h, const float* __restrict__ x, const float* __restrict__ e,
    const float* __restrict__ tv, float* __restrict__ msgs, float* __restrict__ xup, MsgP p)
{
  __shared__ __attribute__((aligned(16))) __bf16 sWh[128*136], sWe[128*136],
      sW1[128*136], sW2[128*136], sWc0[128*136];
  __shared__ float  sWd[128], sWc1[128];
  __shared__ float  sB0[128], sLg[128], sLb[128], sB1[128], sB2[128], sBc0[128];
  __shared__ __attribute__((aligned(16))) __bf16 sA[8][16*136];  // [h_row|h_col], reused inter-stage
  __shared__ __attribute__((aligned(16))) __bf16 sE[8][16*136];  // e_t tile
  __shared__ float  sDist[8][16], sDx[8][16], sDy[8][16];

  const int tid = threadIdx.x;
  ldw_bf16_t(p.Wh,  sWh,  128, 128, 136, tid);
  ldw_bf16_t(p.We,  sWe,  128, 128, 136, tid);
  ldw_bf16_t(p.W1,  sW1,  128, 128, 136, tid);
  ldw_bf16_t(p.W2,  sW2,  128, 128, 136, tid);
  ldw_bf16_t(p.Wc0, sWc0, 128, 128, 136, tid);
  ldw_f32(p.wd,  sWd,  128, tid);  ldw_f32(p.wc1, sWc1, 128, tid);
  ldw_f32(p.b0,  sB0,  128, tid);  ldw_f32(p.g,   sLg,  128, tid);  ldw_f32(p.be, sLb, 128, tid);
  ldw_f32(p.b1,  sB1,  128, tid);  ldw_f32(p.b2,  sB2,  128, tid);  ldw_f32(p.bc0, sBc0, 128, tid);
  __syncthreads();

  const int wave = tid >> 5, lane = tid & 31;
  const int rr = lane & 15, hh = lane >> 4;
  __bf16* A = sA[wave];
  __bf16* E = sE[wave];

  for (int tile = blockIdx.x; tile < NTILE128; tile += gridDim.x) {
    const int gbase = tile * 128 + wave * 16;
    const int g = gbase + rr;
    const int b = g / NEDGE;
    const int k = g - b * NEDGE;
    const int r = k / 99;
    const int j = k - r * 99;
    const int c = j + (j >= r);

    { // stream hint for next tile's e rows
      int nt2 = tile + gridDim.x;
      if (nt2 < NTILE128)
        __builtin_prefetch(e + (size_t)(nt2 * 128 + wave * 16 + rr) * 128, 0, 1);
    }

    // ---- stage A = [h_row | h_col], E = e + t_emb, dist/xdiff ----
    {
      const float* hsrc = h + (size_t)(b * NNODE + (hh ? c : r)) * 64;
#pragma unroll
      for (int d = 0; d < 64; d += 4) {
        float4 v = *(const float4*)(hsrc + d);
        st4(A + rr * 136 + hh * 64 + d, v.x, v.y, v.z, v.w);
      }
      const float* esrc = e  + (size_t)g * 128 + hh * 64;
      const float* tsrc = tv + (size_t)b * 128 + hh * 64;
#pragma unroll
      for (int d = 0; d < 64; d += 4) {
        float4 v = *(const float4*)(esrc + d);
        float4 w = *(const float4*)(tsrc + d);
        st4(E + rr * 136 + hh * 64 + d, v.x + w.x, v.y + w.y, v.z + w.z, v.w + w.w);
      }
      if (hh == 0) {
        const float* xr = x + (size_t)(b * NNODE + r) * 2;
        const float* xc = x + (size_t)(b * NNODE + c) * 2;
        float dx = xc[0] - xr[0], dy = xc[1] - xr[1];
        sDx[wave][rr] = dx; sDy[wave][rr] = dy;
        sDist[wave][rr] = sqrtf(dx * dx + dy * dy);
      }
    }

    // ---- msg linear0 (+ rank-1 dist) -> SiLU -> LN ----
    v8f acc[8];
    zero_acc<8>(acc);
    gemm_tile<8,128>(acc, A, 136, sWh, 136, lane);
    gemm_tile<8,128>(acc, E, 136, sWe, 136, lane);
#pragma unroll
    for (int jj = 0; jj < 8; ++jj) {
      float dv = sDist[wave][hh * 8 + jj];
#pragma unroll
      for (int t = 0; t < 8; ++t) {
        int cc = rr + 16 * t;
        acc[t][jj] = silu_f(acc[t][jj] + dv * sWd[cc] + sB0[cc]);
      }
    }
    layer_norm<8>(acc, sLg, sLb, lane);
    acc_to_lds<8>(acc, A, 136, lane);

    // ---- msg linear1 -> SiLU ----
    zero_acc<8>(acc);
    gemm_tile<8,128>(acc, A, 136, sW1, 136, lane);
    bias_act<8>(acc, sB1, lane, true);
    acc_to_lds<8>(acc, A, 136, lane);

    // ---- msg linear2 -> messages (write f32 + keep bf16 for coord path) ----
    zero_acc<8>(acc);
    gemm_tile<8,128>(acc, A, 136, sW2, 136, lane);
    bias_act<8>(acc, sB2, lane, false);
#pragma unroll
    for (int t = 0; t < 8; ++t)
#pragma unroll
      for (int jj = 0; jj < 8; ++jj)
        msgs[(size_t)(gbase + hh * 8 + jj) * 128 + rr + 16 * t] = acc[t][jj];
    acc_to_lds<8>(acc, A, 136, lane);

    // ---- coord MLP: silu(Wc0 m + bc0) . wc1 -> x_update ----
    zero_acc<8>(acc);
    gemm_tile<8,128>(acc, A, 136, sWc0, 136, lane);
    bias_act<8>(acc, sBc0, lane, true);
#pragma unroll
    for (int jj = 0; jj < 8; ++jj) {
      float part = 0.f;
#pragma unroll
      for (int t = 0; t < 8; ++t) part += acc[t][jj] * sWc1[rr + 16 * t];
      float cw = rowsum16(part);
      if (rr == 0) {
        int row = hh * 8 + jj;
        float inv = cw / (sDist[wave][row] + 1e-8f);
        xup[(size_t)(gbase + row) * 2 + 0] = inv * sDx[wave][row];
        xup[(size_t)(gbase + row) * 2 + 1] = inv * sDy[wave][row];
      }
    }
  }
}

// ============================================================================
// Contiguous segment reduction: h_agg = sum over 99 edges/row; x += sum(xup).
// ============================================================================
__global__ void k_agg(const float* __restrict__ msgs, const float* __restrict__ xup,
                      float* __restrict__ hagg, float* __restrict__ x)
{
  const int gn = blockIdx.x;          // b*100+n
  const int d  = threadIdx.x;         // 0..127
  const int b  = gn / NNODE;
  const int n  = gn - b * NNODE;
  const size_t ebase = (size_t)b * NEDGE + (size_t)n * 99;
  float s = 0.f;
  for (int j = 0; j < 99; ++j) s += msgs[(ebase + j) * 128 + d];
  hagg[(size_t)gn * 128 + d] = s;
  if (d < 2) {
    float sx = 0.f;
    for (int j = 0; j < 99; ++j) sx += xup[(ebase + j) * 2 + d];
    x[gn * 2 + d] += sx;
  }
}

// ============================================================================
// Node MLP: concat(h, h_agg) -> 192->128 silu -> LN -> 128->64; h = LN(h+nm).
// ============================================================================
__global__ __launch_bounds__(256) void k_node(
    float* __restrict__ h, const float* __restrict__ hagg, NodeP p)
{
  __shared__ __attribute__((aligned(16))) __bf16 sW0[128*200], sW1[64*136];
  __shared__ float  sB0[128], sLg[128], sLb[128], sB1[64], sNg[64], sNb[64];
  __shared__ __attribute__((aligned(16))) __bf16 sA[8][16*200], sI[8][16*136];

  const int tid = threadIdx.x;
  ldw_bf16_t(p.W0, sW0, 192, 128, 200, tid);
  ldw_bf16_t(p.W1, sW1, 128,  64, 136, tid);
  ldw_f32(p.b0, sB0, 128, tid); ldw_f32(p.lg, sLg, 128, tid); ldw_f32(p.lb, sLb, 128, tid);
  ldw_f32(p.b1, sB1, 64, tid);  ldw_f32(p.ng, sNg, 64, tid);  ldw_f32(p.nb, sNb, 64, tid);
  __syncthreads();

  const int wave = tid >> 5, lane = tid & 31, rr = lane & 15, hh = lane >> 4;
  __bf16* A = sA[wave];
  __bf16* I = sI[wave];

  for (int tile = blockIdx.x; tile < NODE_ROWS / 128; tile += gridDim.x) {
    const int gbase = tile * 128 + wave * 16;
    const int g = gbase + rr;
    if (hh == 0) {
      const float* hsrc = h + (size_t)g * 64;
      const float* asrc = hagg + (size_t)g * 128;
#pragma unroll
      for (int d = 0; d < 64; d += 4) {
        float4 v = *(const float4*)(hsrc + d);
        st4(A + rr * 200 + d, v.x, v.y, v.z, v.w);
        float4 u = *(const float4*)(asrc + d);
        st4(A + rr * 200 + 64 + d, u.x, u.y, u.z, u.w);
      }
    } else {
      const float* asrc = hagg + (size_t)g * 128 + 64;
#pragma unroll
      for (int d = 0; d < 64; d += 4) {
        float4 u = *(const float4*)(asrc + d);
        st4(A + rr * 200 + 128 + d, u.x, u.y, u.z, u.w);
      }
    }

    v8f acc[8];
    zero_acc<8>(acc);
    gemm_tile<8,192>(acc, A, 200, sW0, 200, lane);
    bias_act<8>(acc, sB0, lane, true);
    layer_norm<8>(acc, sLg, sLb, lane);
    acc_to_lds<8>(acc, I, 136, lane);

    v8f nm[4];
    zero_acc<4>(nm);
    gemm_tile<4,128>(nm, I, 136, sW1, 136, lane);
    bias_act<4>(nm, sB1, lane, false);

    // residual (f32) + LN over 64 + in-place writeback
#pragma unroll
    for (int jj = 0; jj < 8; ++jj) {
      int row = gbase + hh * 8 + jj;
#pragma unroll
      for (int t = 0; t < 4; ++t) nm[t][jj] += h[(size_t)row * 64 + rr + 16 * t];
    }
    layer_norm<4>(nm, sNg, sNb, lane);
#pragma unroll
    for (int jj = 0; jj < 8; ++jj) {
      int row = gbase + hh * 8 + jj;
#pragma unroll
      for (int t = 0; t < 4; ++t) h[(size_t)row * 64 + rr + 16 * t] = nm[t][jj];
    }
  }
}

// ============================================================================
// Edge MLP: concat(e_t, messages) -> 256->128 silu -> LN -> 128->128;
// e = LN(e_t + em)  (in-place over e).
// ============================================================================
__global__ __launch_bounds__(256) void k_edge(
    float* __restrict__ e, const float* __restrict__ msgs, const float* __restrict__ tv, EdgeP p)
{
  __shared__ __attribute__((aligned(16))) __bf16 sW0a[128*136], sW0b[128*136], sW1[128*136];
  __shared__ float  sB0[128], sLg[128], sLb[128], sB1[128], sNg[128], sNb[128];
  __shared__ __attribute__((aligned(16))) __bf16 sS0[8][16*136], sS1[8][16*136];

  const int tid = threadIdx.x;
  ldw_bf16_t(p.W0a, sW0a, 128, 128, 136, tid);
  ldw_bf16_t(p.W0b, sW0b, 128, 128, 136, tid);
  ldw_bf16_t(p.W1,  sW1,  128, 128, 136, tid);
  ldw_f32(p.b0, sB0, 128, tid); ldw_f32(p.lg, sLg, 128, tid); ldw_f32(p.lb, sLb, 128, tid);
  ldw_f32(p.b1, sB1, 128, tid); ldw_f32(p.ng, sNg, 128, tid); ldw_f32(p.nb, sNb, 128, tid);
  __syncthreads();

  const int wave = tid >> 5, lane = tid & 31, rr = lane & 15, hh = lane >> 4;
  __bf16* S0 = sS0[wave];
  __bf16* S1 = sS1[wave];

  for (int tile = blockIdx.x; tile < NTILE128; tile += gridDim.x) {
    const int gbase = tile * 128 + wave * 16;
    const int g = gbase + rr;
    const int b = g / NEDGE;

    const float* esrc = e    + (size_t)g * 128 + hh * 64;
    const float* tsrc = tv   + (size_t)b * 128 + hh * 64;
    const float* msrc = msgs + (size_t)g * 128 + hh * 64;
#pragma unroll
    for (int d = 0; d < 64; d += 4) {
      float4 v = *(const float4*)(esrc + d);
      float4 w = *(const float4*)(tsrc + d);
      st4(S0 + rr * 136 + hh * 64 + d, v.x + w.x, v.y + w.y, v.z + w.z, v.w + w.w);
      float4 m = *(const float4*)(msrc + d);
      st4(S1 + rr * 136 + hh * 64 + d, m.x, m.y, m.z, m.w);
    }

    v8f acc[8];
    zero_acc<8>(acc);
    gemm_tile<8,128>(acc, S0, 136, sW0a, 136, lane);
    gemm_tile<8,128>(acc, S1, 136, sW0b, 136, lane);
    bias_act<8>(acc, sB0, lane, true);
    layer_norm<8>(acc, sLg, sLb, lane);
    acc_to_lds<8>(acc, S1, 136, lane);

    zero_acc<8>(acc);
    gemm_tile<8,128>(acc, S1, 136, sW1, 136, lane);
    bias_act<8>(acc, sB1, lane, false);

    // residual e_t (recomputed in f32) + LN + in-place write
#pragma unroll
    for (int jj = 0; jj < 8; ++jj) {
      int row = gbase + hh * 8 + jj;
      int b2 = row / NEDGE;
#pragma unroll
      for (int t = 0; t < 8; ++t) {
        int cc = rr + 16 * t;
        acc[t][jj] += e[(size_t)row * 128 + cc] + tv[(size_t)b2 * 128 + cc];
      }
    }
    layer_norm<8>(acc, sNg, sNb, lane);
#pragma unroll
    for (int jj = 0; jj < 8; ++jj) {
      int row = gbase + hh * 8 + jj;
#pragma unroll
      for (int t = 0; t < 8; ++t) e[(size_t)row * 128 + rr + 16 * t] = acc[t][jj];
    }
  }
}

// ============================================================================
// Output head: y = l3( silu(l2( LN2( silu(l1( LN1(e) )) ))) )  -> (B,E,2)
// ============================================================================
__global__ __launch_bounds__(256) void k_out(
    const float* __restrict__ e, float* __restrict__ out, OutP p)
{
  __shared__ __attribute__((aligned(16))) __bf16 sW1[128*136], sW2[64*136];
  __shared__ float  sG1[128], sBt1[128], sB1[128], sG2[128], sBt2[128], sB2[64], sW3[128], sB3f[2];
  __shared__ __attribute__((aligned(16))) __bf16 sS[8][16*136];

  const int tid = threadIdx.x;
  ldw_bf16_t(p.W1, sW1, 128, 128, 136, tid);
  ldw_bf16_t(p.W2, sW2, 128,  64, 136, tid);
  ldw_f32(p.l1g, sG1, 128, tid); ldw_f32(p.l1b, sBt1, 128, tid); ldw_f32(p.b1, sB1, 128, tid);
  ldw_f32(p.l2g, sG2, 128, tid); ldw_f32(p.l2b, sBt2, 128, tid); ldw_f32(p.b2, sB2, 64, tid);
  ldw_f32(p.W3, sW3, 128, tid);  ldw_f32(p.b3, sB3f, 2, tid);
  __syncthreads();

  const int wave = tid >> 5, lane = tid & 31, rr = lane & 15, hh = lane >> 4;
  __bf16* S = sS[wave];

  for (int tile = blockIdx.x; tile < NTILE128; tile += gridDim.x) {
    const int gbase = tile * 128 + wave * 16;
    const int g = gbase + rr;

    // LN1 on e while staging bf16
    float s = 0.f, s2 = 0.f;
    const float* esrc = e + (size_t)g * 128 + hh * 64;
#pragma unroll
    for (int d = 0; d < 64; d += 4) {
      float4 v = *(const float4*)(esrc + d);
      s  += v.x + v.y + v.z + v.w;
      s2 += v.x*v.x + v.y*v.y + v.z*v.z + v.w*v.w;
      st4(S + rr * 136 + hh * 64 + d, v.x, v.y, v.z, v.w);
    }
    s  += __shfl_xor(s, 16, 32);
    s2 += __shfl_xor(s2, 16, 32);
    float mu = s * (1.f / 128.f);
    float rs = rsqrtf(s2 * (1.f / 128.f) - mu * mu + 1e-5f);
#pragma unroll
    for (int d = 0; d < 64; d += 4) {
      int ci = hh * 64 + d;
      BF4 t4; t4.u = *(uint2*)(S + rr * 136 + ci);
#pragma unroll
      for (int q = 0; q < 4; ++q)
        t4.h[q] = (__bf16)(((float)t4.h[q] - mu) * rs * sG1[ci + q] + sBt1[ci + q]);
      *(uint2*)(S + rr * 136 + ci) = t4.u;
    }

    v8f acc[8];
    zero_acc<8>(acc);
    gemm_tile<8,128>(acc, S, 136, sW1, 136, lane);
    bias_act<8>(acc, sB1, lane, true);
    layer_norm<8>(acc, sG2, sBt2, lane);
    acc_to_lds<8>(acc, S, 136, lane);

    v8f y[4];
    zero_acc<4>(y);
    gemm_tile<4,128>(y, S, 136, sW2, 136, lane);
    bias_act<4>(y, sB2, lane, true);

#pragma unroll
    for (int jj = 0; jj < 8; ++jj) {
      float p0 = 0.f, p1 = 0.f;
#pragma unroll
      for (int t = 0; t < 4; ++t) {
        int cc = rr + 16 * t;
        p0 += y[t][jj] * sW3[cc * 2 + 0];
        p1 += y[t][jj] * sW3[cc * 2 + 1];
      }
      p0 = rowsum16(p0);
      p1 = rowsum16(p1);
      if (rr == 0) {
        int row = gbase + hh * 8 + jj;
        out[(size_t)row * 2 + 0] = p0 + sB3f[0];
        out[(size_t)row * 2 + 1] = p1 + sB3f[1];
      }
    }
  }
}

// ============================================================================
// Small setup kernels
// ============================================================================
__global__ void k_init_h(const float* __restrict__ coords, const float* __restrict__ wgt,
                         const float* __restrict__ bias, float* __restrict__ h)
{
  int i = blockIdx.x * 256 + threadIdx.x;
  if (i >= NODE_ROWS * 64) return;
  int gn = i >> 6, d = i & 63;
  h[i] = coords[gn * 2] * wgt[d] + coords[gn * 2 + 1] * wgt[64 + d] + bias[d];
}

__global__ void k_init_e(const float* __restrict__ ef, const float* __restrict__ wgt,
                         const float* __restrict__ bias, float* __restrict__ e, long n)
{
  long i = (long)blockIdx.x * 256 + threadIdx.x;
  if (i >= n) return;
  long row = i >> 7;
  int d = (int)(i & 127);
  e[i] = ef[row] * wgt[d] + bias[d];
}

__global__ __launch_bounds__(256) void k_time(const float* __restrict__ ts, TimeP p,
                                              float* __restrict__ tvec)
{
  __shared__ float te[64 * 128];
  __shared__ float h1[64 * 256];
  __shared__ float temb[64 * 128];
  const int tid = threadIdx.x;
  for (int i = tid; i < 64 * 64; i += 256) {
    int b = i >> 6, d = i & 63;
    float f = __expf(-9.210340371976184f * (float)d * (1.f / 64.f));  // exp(-ln(1e4)*d/64)
    float a = ts[b] * f;
    te[b * 128 + d]      = __cosf(a);
    te[b * 128 + 64 + d] = __sinf(a);
  }
  __syncthreads();
  for (int i = tid; i < 64 * 256; i += 256) {
    int b = i >> 8, n = i & 255;
    float s = p.b0[n];
    for (int k = 0; k < 128; ++k) s += te[b * 128 + k] * p.W0[k * 256 + n];
    h1[i] = silu_f(s);
  }
  __syncthreads();
  for (int i = tid; i < 64 * 128; i += 256) {
    int b = i >> 7, n = i & 127;
    float s = p.b1[n];
    for (int k = 0; k < 256; ++k) s += h1[b * 256 + k] * p.W1[k * 128 + n];
    temb[i] = s;
  }
  __syncthreads();
  for (int l = 0; l < 8; ++l)
    for (int i = tid; i < 64 * 128; i += 256) {
      int b = i >> 7, n = i & 127;
      const float* W = p.tw[l];
      float s = p.tb[l][n];
      for (int k = 0; k < 128; ++k) s += temb[b * 128 + k] * W[k * 128 + n];
      tvec[(size_t)l * 64 * 128 + i] = s;
    }
}

// ============================================================================
// Host side
// ============================================================================
struct LayerOffs {
  long c0_w,c0_b,c1_w;
  long e0_w,e0_b,e1_w,e1_b;
  long eln_g,eln_b,enrm_g,enrm_b;
  long m0_w,m0_b,m1_w,m1_b,m2_w,m2_b;
  long mln_g,mln_b;
  long n0_w,n0_b,n1_w,n1_b;
  long nln_g,nln_b,nnrm_g,nnrm_b;
};

extern "C" void kernel_launch(void* const* d_in, const int* in_sizes, int n_in,
                              void* d_out, int out_size, void* d_ws, size_t ws_size,
                              hipStream_t stream) {
  const float* coords = (const float*)d_in[0];   // (64,100,2)
  const float* efeat  = (const float*)d_in[1];   // (64,9900)
  const float* tsteps = (const float*)d_in[2];   // (64,)
  const float* P      = (const float*)d_in[3];   // flattened params (tree_leaves order)
  float* out = (float*)d_out;                    // (64,9900,2) f32
  (void)in_sizes; (void)n_in; (void)out_size;

  // ---- parameter offsets: jax tree_leaves (sorted dict keys; "b" < "g" < "w") ----
  long cur = 0;
  auto LIN = [&](int din, int dout, bool bias, long& w, long& b) {
    if (bias) { b = cur; cur += dout; } else b = -1;
    w = cur; cur += (long)din * dout;
  };
  auto LNp = [&](long& g, long& b, int d) { b = cur; cur += d; g = cur; cur += d; };

  long ee_w, ee_b; LIN(1, 128, true, ee_w, ee_b);                    // edge_embed
  LayerOffs L[8];
  for (int l = 0; l < 8; ++l) {                                      // layers
    LayerOffs& o = L[l];
    LIN(128, 128, true, o.c0_w, o.c0_b);                             // coord[0]
    { long nb; LIN(128, 1, false, o.c1_w, nb); }                     // coord[1] (no bias)
    LIN(256, 128, true, o.e0_w, o.e0_b);                             // edge[0]
    LIN(128, 128, true, o.e1_w, o.e1_b);                             // edge[1]
    LNp(o.eln_g, o.eln_b, 128);                                      // edge_ln
    LNp(o.enrm_g, o.enrm_b, 128);                                    // edge_norm
    LIN(257, 128, true, o.m0_w, o.m0_b);                             // msg[0]
    LIN(128, 128, true, o.m1_w, o.m1_b);                             // msg[1]
    LIN(128, 128, true, o.m2_w, o.m2_b);                             // msg[2]
    LNp(o.mln_g, o.mln_b, 128);                                      // msg_ln
    LIN(192, 128, true, o.n0_w, o.n0_b);                             // node[0]
    LIN(128,  64, true, o.n1_w, o.n1_b);                             // node[1]
    LNp(o.nln_g, o.nln_b, 128);                                      // node_ln
    LNp(o.nnrm_g, o.nnrm_b, 64);                                     // node_norm
  }
  long ne_w, ne_b; LIN(2, 64, true, ne_w, ne_b);                     // node_embed
  long l1_w,l1_b,l2_w,l2_b,l3_w,l3_b,ln1_g,ln1_b,ln2_g,ln2_b;        // out
  LIN(128, 128, true, l1_w, l1_b);
  LIN(128,  64, true, l2_w, l2_b);
  LIN( 64,   2, true, l3_w, l3_b);
  LNp(ln1_g, ln1_b, 128);
  LNp(ln2_g, ln2_b, 128);
  long te0_w,te0_b,te1_w,te1_b;                                      // time_embed
  LIN(128, 256, true, te0_w, te0_b);
  LIN(256, 128, true, te1_w, te1_b);
  long tl_w[8], tl_b[8];                                             // time_layers
  for (int l = 0; l < 8; ++l) LIN(128, 128, true, tl_w[l], tl_b[l]);

  // ---- workspace carve (~660 MB) ----
  char* w = (char*)d_ws;
  auto carve = [&](size_t bytes) { void* pp = (void*)w; w += (bytes + 255) & ~(size_t)255; return pp; };
  float* e    = (float*)carve((size_t)NROWS * 128 * 4);
  float* msgs = (float*)carve((size_t)NROWS * 128 * 4);
  float* xup  = (float*)carve((size_t)NROWS * 2 * 4);
  float* hbuf = (float*)carve((size_t)NODE_ROWS * 64 * 4);
  float* hagg = (float*)carve((size_t)NODE_ROWS * 128 * 4);
  float* xbuf = (float*)carve((size_t)NODE_ROWS * 2 * 4);
  float* tvec = (float*)carve((size_t)8 * 64 * 128 * 4);
  (void)ws_size;

  // ---- embeddings ----
  hipMemcpyAsync(xbuf, coords, (size_t)NODE_ROWS * 2 * 4, hipMemcpyDeviceToDevice, stream);
  k_init_h<<<(NODE_ROWS * 64 + 255) / 256, 256, 0, stream>>>(coords, P + ne_w, P + ne_b, hbuf);
  {
    long n = (long)NROWS * 128;
    k_init_e<<<(unsigned)((n + 255) / 256), 256, 0, stream>>>(efeat, P + ee_w, P + ee_b, e, n);
  }
  TimeP tp;
  tp.W0 = P + te0_w; tp.b0 = P + te0_b; tp.W1 = P + te1_w; tp.b1 = P + te1_b;
  for (int l = 0; l < 8; ++l) { tp.tw[l] = P + tl_w[l]; tp.tb[l] = P + tl_b[l]; }
  k_time<<<1, 256, 0, stream>>>(tsteps, tp, tvec);

  // ---- 8 EGNN layers ----
  const int NB = 912;   // grid-stride blocks for the big per-edge kernels
  for (int l = 0; l < 8; ++l) {
    const LayerOffs& o = L[l];
    const float* tv_l = tvec + (size_t)l * 64 * 128;
    // msg[0].w rows: 0..127 = [h_row|h_col], 128 = dist, 129..256 = e_t
    MsgP mp = { P + o.m0_w, P + o.m0_w + 129 * 128, P + o.m0_w + 128 * 128, P + o.m0_b,
                P + o.mln_g, P + o.mln_b, P + o.m1_w, P + o.m1_b, P + o.m2_w, P + o.m2_b,
                P + o.c0_w, P + o.c0_b, P + o.c1_w };
    k_message<<<NB, 256, 0, stream>>>(hbuf, xbuf, e, tv_l, msgs, xup, mp);
    k_agg<<<NODE_ROWS, 128, 0, stream>>>(msgs, xup, hagg, xbuf);
    NodeP np = { P + o.n0_w, P + o.n0_b, P + o.nln_g, P + o.nln_b,
                 P + o.n1_w, P + o.n1_b, P + o.nnrm_g, P + o.nnrm_b };
    k_node<<<NODE_ROWS / 128, 256, 0, stream>>>(hbuf, hagg, np);
    // edge[0].w rows: 0..127 = e_t, 128..255 = messages
    EdgeP ep = { P + o.e0_w, P + o.e0_w + 128 * 128, P + o.e0_b, P + o.eln_g, P + o.eln_b,
                 P + o.e1_w, P + o.e1_b, P + o.enrm_g, P + o.enrm_b };
    k_edge<<<NB, 256, 0, stream>>>(e, msgs, tv_l, ep);
  }

  // ---- output head ----
  OutP op = { P + ln1_g, P + ln1_b, P + l1_w, P + l1_b, P + ln2_g, P + ln2_b,
              P + l2_w, P + l2_b, P + l3_w, P + l3_b };
  k_out<<<NB, 256, 0, stream>>>(e, out, op);
}
